// LocalTransformer_8718783611581
// MI455X (gfx1250) — matrix operand order repaired and checked by reading, not compiled, not run on hardware
//
#include <hip/hip_runtime.h>
#include <hip/hip_bf16.h>

typedef _Float16 h16 __attribute__((ext_vector_type(16)));
typedef _Float16 h8  __attribute__((ext_vector_type(8)));
typedef float    f8  __attribute__((ext_vector_type(8)));

#define B_    8
#define L_    4096
#define C_    512
#define W_    20
#define HOP_  10
#define S_    408          // arange(0, 4096-20+1, 10) -> 408 windows
#define E3_   1536
#define LNEPS 1e-5f

#define SA_PITCH    520                    // halves per row of normed A (pad vs bank conflicts)
#define SQKV_OFF    (32 * SA_PITCH)        // half offset of qkv buffer
#define SMEM_HALVES (SQKV_OFF + 32 * E3_)  // 16640 + 49152 halves = 131584 bytes

#define WIN_HALVES  (E3_ * C_)             // 786432 f16 for w_in in d_ws

// ---------------- weight f32 -> f16 conversion ----------------
__global__ __launch_bounds__(256)
void prep_weights(const float* __restrict__ wi, const float* __restrict__ wo,
                  _Float16* __restrict__ ws)
{
    int i = blockIdx.x * 256 + threadIdx.x;
    if (i < WIN_HALVES) {
        ws[i] = (_Float16)wi[i];
    } else {
        int j = i - WIN_HALVES;
        if (j < C_ * C_) ws[WIN_HALVES + j] = (_Float16)wo[j];
    }
}

// ---------------- per-segment fused LN + MHA kernel ----------------
__global__ __launch_bounds__(256)
void seg_kernel(const float* __restrict__ x,
                const float* __restrict__ ln_g, const float* __restrict__ ln_b,
                const _Float16* __restrict__ wInH, const float* __restrict__ b_in,
                const _Float16* __restrict__ wOutH, const float* __restrict__ b_out,
                float* __restrict__ out)
{
    extern __shared__ __align__(16) _Float16 smem[];
    _Float16* sA   = smem;             // 32 x 520 f16 : normed A; reused as score buf
    _Float16* sQKV = smem + SQKV_OFF;  // 32 x 1536 f16 : q|k|v; q region reused as attn_out

    const int seg  = blockIdx.x;
    const int b    = seg / S_;
    const int s    = seg - b * S_;
    const int tid  = threadIdx.x;
    const int wave = tid >> 5;
    const int lane = tid & 31;
    const int lm   = lane & 15;
    const int kbA  = (lane < 16) ? 0 : 16;  // A-frag K base: contiguous 16
    const int kbB  = (lane < 16) ? 0 : 16;  // B-frag K base: contiguous 16
    const int rOff = (lane < 16) ? 0 : 8;   // C-frag row offset per lane half

    // ---- Phase 0: LayerNorm rows -> sA f16 (rows 20..31 zeroed) ----
    {
        const int row = tid >> 3;   // 0..31
        const int sub = tid & 7;    // 8 lanes per row, 64 cols each
        if (row < W_) {
            const float4* xv = (const float4*)(x + (size_t)(b * L_ + s * HOP_ + row) * C_)
                               + sub * 16;
            float sum = 0.f, sq = 0.f;
            float4 buf[16];
#pragma unroll
            for (int i = 0; i < 16; ++i) {
                float4 v = xv[i]; buf[i] = v;
                sum += v.x + v.y + v.z + v.w;
                sq  += v.x * v.x + v.y * v.y + v.z * v.z + v.w * v.w;
            }
#pragma unroll
            for (int m = 1; m <= 4; m <<= 1) {
                sum += __shfl_xor(sum, m, 32);
                sq  += __shfl_xor(sq,  m, 32);
            }
            const float mean = sum * (1.0f / C_);
            const float var  = sq * (1.0f / C_) - mean * mean;   // biased, matches jnp.var
            const float rstd = rsqrtf(var + LNEPS);
            _Float16* dst = sA + row * SA_PITCH + sub * 64;
            const float* g  = ln_g + sub * 64;
            const float* bb = ln_b + sub * 64;
#pragma unroll
            for (int i = 0; i < 16; ++i) {
                float4 v = buf[i];
                dst[i * 4 + 0] = (_Float16)((v.x - mean) * rstd * g[i * 4 + 0] + bb[i * 4 + 0]);
                dst[i * 4 + 1] = (_Float16)((v.y - mean) * rstd * g[i * 4 + 1] + bb[i * 4 + 1]);
                dst[i * 4 + 2] = (_Float16)((v.z - mean) * rstd * g[i * 4 + 2] + bb[i * 4 + 2]);
                dst[i * 4 + 3] = (_Float16)((v.w - mean) * rstd * g[i * 4 + 3] + bb[i * 4 + 3]);
            }
        } else {
            _Float16* dst = sA + row * SA_PITCH + sub * 64;
            h8 z = {};
#pragma unroll
            for (int i = 0; i < 8; ++i) ((h8*)dst)[i] = z;
        }
    }
    __syncthreads();

    // ---- Phase 1: QKV GEMM (32x1536 = A(32x512) x w_in^T), K=512 ----
#pragma unroll 1
    for (int t = wave; t < 192; t += 8) {
        const int mt = t & 1;       // M tile 0..1
        const int nt = t >> 1;      // N tile 0..95
        f8 acc = {};
        const _Float16* arow = sA + (mt * 16 + lm) * SA_PITCH;
        const _Float16* brow = wInH + (size_t)(nt * 16 + lm) * C_;
#pragma unroll
        for (int kk = 0; kk < 16; ++kk) {
            const int k0 = kk * 32;
            h16 af, bf;
            ((h8*)&af)[0] = *(const h8*)(arow + k0 + kbA);
            ((h8*)&af)[1] = *(const h8*)(arow + k0 + kbA + 8);
            ((h8*)&bf)[0] = *(const h8*)(brow + k0 + kbB);
            ((h8*)&bf)[1] = *(const h8*)(brow + k0 + kbB + 8);
            acc = __builtin_amdgcn_wmma_f32_16x16x32_f16(false, af, false, bf,
                                                         (short)0, acc, false, false);
        }
        const int n = nt * 16 + lm;
        const float bias = b_in[n];
        _Float16* d = sQKV + (mt * 16 + rOff) * E3_ + n;
#pragma unroll
        for (int r = 0; r < 8; ++r) d[r * E3_] = (_Float16)(acc[r] + bias);
    }
    __syncthreads();

    // ---- Phase 2a: scores = q k^T / 8 (wave handles head=wave). sA reused as f32 buf ----
    float* sS = (float*)smem + wave * 1024;   // 32x32 f32 per wave (fits in dead sA)
    {
        const int h = wave;
#pragma unroll
        for (int mt = 0; mt < 2; ++mt) {
#pragma unroll
            for (int nt = 0; nt < 2; ++nt) {
                f8 acc = {};
                const _Float16* qrow = sQKV + (mt * 16 + lm) * E3_ + h * 64;
                const _Float16* krow = sQKV + (nt * 16 + lm) * E3_ + 512 + h * 64;
#pragma unroll
                for (int kk = 0; kk < 2; ++kk) {
                    const int k0 = kk * 32;
                    h16 af, bf;
                    ((h8*)&af)[0] = *(const h8*)(qrow + k0 + kbA);
                    ((h8*)&af)[1] = *(const h8*)(qrow + k0 + kbA + 8);
                    ((h8*)&bf)[0] = *(const h8*)(krow + k0 + kbB);
                    ((h8*)&bf)[1] = *(const h8*)(krow + k0 + kbB + 8);
                    acc = __builtin_amdgcn_wmma_f32_16x16x32_f16(false, af, false, bf,
                                                                 (short)0, acc, false, false);
                }
                float* d = sS + (mt * 16 + rOff) * 32 + nt * 16 + lm;
#pragma unroll
                for (int r = 0; r < 8; ++r) d[r * 32] = acc[r] * 0.125f;  // 1/sqrt(64)
            }
        }
    }
    __syncthreads();

    // ---- Phase 2b: row softmax over keys 0..19, write f16 probs in place ----
    {
        _Float16* probs = (_Float16*)sS;   // row stride 64 halves (=128B, same as f32 row)
        const int m = lane;
        _Float16* pr = probs + m * 64;
        if (m < W_) {
            float sc[W_];
            float mx = -3.4e38f;
#pragma unroll
            for (int k = 0; k < W_; ++k) { sc[k] = sS[m * 32 + k]; mx = fmaxf(mx, sc[k]); }
            float sum = 0.f;
#pragma unroll
            for (int k = 0; k < W_; ++k) { sc[k] = __expf(sc[k] - mx); sum += sc[k]; }
            const float inv = 1.0f / sum;
#pragma unroll
            for (int k = 0; k < W_; ++k) pr[k] = (_Float16)(sc[k] * inv);
#pragma unroll
            for (int k = W_; k < 32; ++k) pr[k] = (_Float16)0.f;     // mask padded keys
        } else {
#pragma unroll
            for (int k = 0; k < 32; ++k) pr[k] = (_Float16)0.f;      // padded query rows
        }
    }
    __syncthreads();

    // ---- Phase 2c: attn_out = probs x V  (K=32 single step); write over q region ----
    {
        const int h = wave;
        const _Float16* probs = (const _Float16*)sS;
#pragma unroll
        for (int mt = 0; mt < 2; ++mt) {
#pragma unroll
            for (int nt = 0; nt < 4; ++nt) {
                f8 acc = {};
                h16 af, bf;
                const _Float16* prow = probs + (mt * 16 + lm) * 64;
                ((h8*)&af)[0] = *(const h8*)(prow + kbA);
                ((h8*)&af)[1] = *(const h8*)(prow + kbA + 8);
                const int col = 1024 + h * 64 + nt * 16 + lm;   // V column for this lane
#pragma unroll
                for (int j = 0; j < 16; ++j) bf[j] = sQKV[(kbB + j) * E3_ + col];
                acc = __builtin_amdgcn_wmma_f32_16x16x32_f16(false, af, false, bf,
                                                             (short)0, acc, false, false);
                _Float16* d = sQKV + (mt * 16 + rOff) * E3_ + h * 64 + nt * 16 + lm;
#pragma unroll
                for (int r = 0; r < 8; ++r) d[r * E3_] = (_Float16)acc[r];
            }
        }
    }
    __syncthreads();

    // ---- Phase 3: out-proj (32x512 = attn_out(32x512) x w_out^T) + bias, atomic add ----
#pragma unroll 1
    for (int t = wave; t < 64; t += 8) {
        const int mt = t & 1;
        const int nt = t >> 1;    // 0..31
        f8 acc = {};
        const _Float16* arow = sQKV + (mt * 16 + lm) * E3_;
        const _Float16* brow = wOutH + (size_t)(nt * 16 + lm) * C_;
#pragma unroll
        for (int kk = 0; kk < 16; ++kk) {
            const int k0 = kk * 32;
            h16 af, bf;
            ((h8*)&af)[0] = *(const h8*)(arow + k0 + kbA);
            ((h8*)&af)[1] = *(const h8*)(arow + k0 + kbA + 8);
            ((h8*)&bf)[0] = *(const h8*)(brow + k0 + kbB);
            ((h8*)&bf)[1] = *(const h8*)(brow + k0 + kbB + 8);
            acc = __builtin_amdgcn_wmma_f32_16x16x32_f16(false, af, false, bf,
                                                         (short)0, acc, false, false);
        }
        const int n = nt * 16 + lm;
        const float bias = b_out[n];
        const int rbase = mt * 16 + rOff;
#pragma unroll
        for (int r = 0; r < 8; ++r) {
            const int m = rbase + r;
            if (m < W_) {
                float* dst = out + (size_t)(b * L_ + s * HOP_ + m) * C_ + n;
                __hip_atomic_fetch_add(dst, acc[r] + bias,
                                       __ATOMIC_RELAXED, __HIP_MEMORY_SCOPE_AGENT);
            }
        }
    }
}

// ---------------- finalize: out = acc/cnt + LN(x) + x (cnt analytic) ----------------
__global__ __launch_bounds__(256)
void finalize_kernel(const float* __restrict__ x, const float* __restrict__ ln_g,
                     const float* __restrict__ ln_b, float* __restrict__ out)
{
    const int wave = threadIdx.x >> 5;
    const int lane = threadIdx.x & 31;
    const int rid  = blockIdx.x * 8 + wave;     // row over B*L
    const int p    = rid & (L_ - 1);
    const size_t base = (size_t)rid * C_;
    const float* xr = x + base;
    float* orow = out + base;

    float cnt;
    if (p >= 4090)                 cnt = 0.f;   // uncovered tail
    else if (p < HOP_ || p >= 4080) cnt = 1.f;  // half-overlap edges
    else                            cnt = 2.f;  // interior: exactly 2 windows

    const float4* xv = (const float4*)xr + lane * 4;
    float4 buf[4];
    float sum = 0.f, sq = 0.f;
#pragma unroll
    for (int i = 0; i < 4; ++i) {
        float4 v = xv[i]; buf[i] = v;
        sum += v.x + v.y + v.z + v.w;
        sq  += v.x * v.x + v.y * v.y + v.z * v.z + v.w * v.w;
    }
#pragma unroll
    for (int m = 1; m <= 16; m <<= 1) {
        sum += __shfl_xor(sum, m, 32);
        sq  += __shfl_xor(sq,  m, 32);
    }
    const float mean = sum * (1.0f / C_);
    const float var  = sq * (1.0f / C_) - mean * mean;
    const float rstd = rsqrtf(var + LNEPS);
    const float invc = (cnt > 0.f) ? (1.0f / cnt) : 0.f;
    const int c0 = lane * 16;
#pragma unroll
    for (int i = 0; i < 4; ++i) {
        const float vv[4] = { buf[i].x, buf[i].y, buf[i].z, buf[i].w };
#pragma unroll
        for (int j = 0; j < 4; ++j) {
            const int c = c0 + i * 4 + j;
            const float xvv = vv[j];
            const float ln = (cnt > 0.f) ? ((xvv - mean) * rstd * ln_g[c] + ln_b[c]) : 0.f;
            orow[c] = orow[c] * invc + ln + xvv;
        }
    }
}

extern "C" void kernel_launch(void* const* d_in, const int* in_sizes, int n_in,
                              void* d_out, int out_size, void* d_ws, size_t ws_size,
                              hipStream_t stream)
{
    (void)in_sizes; (void)n_in; (void)ws_size;
    const float* x     = (const float*)d_in[0];
    const float* ln_g  = (const float*)d_in[1];
    const float* ln_b  = (const float*)d_in[2];
    const float* w_in  = (const float*)d_in[3];
    const float* b_in  = (const float*)d_in[4];
    const float* w_out = (const float*)d_in[5];
    const float* b_out = (const float*)d_in[6];
    float* out = (float*)d_out;
    _Float16* wH = (_Float16*)d_ws;

    hipMemsetAsync(d_out, 0, (size_t)out_size * sizeof(float), stream);

    prep_weights<<<(WIN_HALVES + C_ * C_) / 256, 256, 0, stream>>>(w_in, w_out, wH);

    seg_kernel<<<B_ * S_, 256, SMEM_HALVES * sizeof(_Float16), stream>>>(
        x, ln_g, ln_b, wH, b_in, wH + WIN_HALVES, b_out, out);

    finalize_kernel<<<(B_ * L_) / 8, 256, 0, stream>>>(x, ln_g, ln_b, out);
}
